// CompositionalResidualMLP_79001628442948
// MI455X (gfx1250) — matrix operand order repaired and checked by reading, not compiled, not run on hardware
//
#include <hip/hip_runtime.h>

// CDNA5 / gfx1250 grouped-GEMM implementation of the routed (mixture-of-experts)
// residual MLP. Rows are bucketed by expert (atomic histogram + prefix sum +
// scatter perm, padded to 16-row tiles), then dense f16 WMMA
// (v_wmma_f32_16x16x32_f16, f32 accumulate) runs on expert-uniform tiles.
// Weights pre-transposed to [expert][N][K] f16 -> B fragment = one contiguous
// 32B load per lane; activations f16 row-major -> A fragment = two contiguous
// 16B loads per lane (contiguous k = 16*(lane/16) + s, same in A and B). Padding lanes read a
// zero-filled scratch row so the inner loop has no per-element selects.

typedef __attribute__((ext_vector_type(16))) _Float16 v16h;
typedef __attribute__((ext_vector_type(8)))  _Float16 v8h;
typedef __attribute__((ext_vector_type(8)))  float    v8f;

#define BATCH    32768
#define INW      144      // D0+D1+2M
#define NEXP     8
#define MAXTILES 2056     // B/16 + NEXP padding tiles
#define MAXSLOTS (MAXTILES * 16)
#define ZROWLEN  384      // max K across layers

// ---------------- init: zero counters + zero-row, poison perm slots ----------------
__global__ void k_init(int* __restrict__ cnt, int* __restrict__ perm0, int* __restrict__ perm1,
                       _Float16* __restrict__ zrow) {
  int i = blockIdx.x * blockDim.x + threadIdx.x;
  if (i < 16) cnt[i] = 0;
  if (i < ZROWLEN) zrow[i] = (_Float16)0.0f;
  if (i < MAXSLOTS) { perm0[i] = -1; perm1[i] = -1; }
}

// ---------------- features fp32(row stride 144) -> f16 [B][128] ----------------
__global__ void k_feat(const float* __restrict__ in, _Float16* __restrict__ out) {
  int i = blockIdx.x * blockDim.x + threadIdx.x;
  if (i >= BATCH * 128) return;
  int r = i >> 7, c = i & 127;
  out[i] = (_Float16)in[(size_t)r * INW + c];
}

// ---------------- weights [M][K][N] fp32 -> [M][N][K] f16 ----------------
__global__ void k_wconv(const float* __restrict__ in, _Float16* __restrict__ out, int K, int N) {
  int i = blockIdx.x * blockDim.x + threadIdx.x;
  int total = NEXP * K * N;
  if (i >= total) return;
  int n = i % N;
  int k = (i / N) % K;
  int m = i / (N * K);
  out[((size_t)m * N + n) * K + k] = (_Float16)in[i];
}

// ---------------- routing: argmax over one-hot + atomic rank ----------------
__global__ void k_route(const float* __restrict__ in, int* __restrict__ e0a, int* __restrict__ e1a,
                        int* __restrict__ r0a, int* __restrict__ r1a, int* __restrict__ cnt) {
  int r = blockIdx.x * blockDim.x + threadIdx.x;
  if (r >= BATCH) return;
  const float* p = in + (size_t)r * INW + 128;
  int e0 = 0, e1 = 0;
  float b0 = p[0], b1 = p[8];
  for (int i = 1; i < 8; ++i) {
    float v0 = p[i], v1 = p[8 + i];
    if (v0 > b0) { b0 = v0; e0 = i; }
    if (v1 > b1) { b1 = v1; e1 = i; }
  }
  e0a[r] = e0; e1a[r] = e1;
  r0a[r] = atomicAdd(&cnt[e0], 1);
  r1a[r] = atomicAdd(&cnt[8 + e1], 1);
}

// ---------------- prefix sum of padded tile counts (tiny, 1 thread) ----------------
__global__ void k_offsets(const int* __restrict__ cnt, int* __restrict__ s0, int* __restrict__ s1) {
  if (threadIdx.x == 0 && blockIdx.x == 0) {
    int a = 0, b = 0;
    s0[0] = 0; s1[0] = 0;
    for (int e = 0; e < 8; ++e) {
      a += (cnt[e] + 15) >> 4;     s0[e + 1] = a;
      b += (cnt[8 + e] + 15) >> 4; s1[e + 1] = b;
    }
  }
}

// ---------------- scatter permutation (row -> padded expert-grouped slot) ----------------
__global__ void k_scatter(const int* __restrict__ e0a, const int* __restrict__ e1a,
                          const int* __restrict__ r0a, const int* __restrict__ r1a,
                          const int* __restrict__ s0, const int* __restrict__ s1,
                          int* __restrict__ perm0, int* __restrict__ perm1) {
  int r = blockIdx.x * blockDim.x + threadIdx.x;
  if (r >= BATCH) return;
  perm0[s0[e0a[r]] * 16 + r0a[r]] = r;
  perm1[s1[e1a[r]] * 16 + r1a[r]] = r;
}

// ---------------- grouped GEMM ----------------
// 8 independent waves per 256-thread block; each wave computes a 16-row x
// (NT*16)-col strip of one expert-uniform tile. K, N, NT are compile-time.
// A: f16 row-major gathered via perm; Wt: f16 [expert][N][K]; bias fp32 [e][N].
template<int K, int N, int NT, bool RELU, bool F32OUT>
__global__ void __launch_bounds__(256)
k_gemm(const _Float16* __restrict__ A, int lda, int aOff,
       const _Float16* __restrict__ Wt, const float* __restrict__ bias,
       const int* __restrict__ perm, const int* __restrict__ starts,
       const _Float16* __restrict__ zrow,
       _Float16* __restrict__ outH, float* __restrict__ outF,
       int ldo, int oOff)
{
  constexpr int NSTRIP = N / (NT * 16);       // col strips per row tile
  const int gw    = blockIdx.x * 8 + (threadIdx.x >> 5);   // global wave id
  const int tile  = gw / NSTRIP;
  const int strip = gw % NSTRIP;
  if (tile >= starts[8]) return;              // beyond last padded group

  int e = 0;
#pragma unroll
  for (int i = 1; i < 8; ++i) if (tile >= starts[i]) e = i;   // owning expert

  const int lane  = threadIdx.x & 31;
  const int col16 = lane & 15;                // A-row idx == B/D-col idx in tile
  const int hi16  = lane >> 4;

  const int rowA = perm[tile * 16 + col16];
  int rowsOut[8];
#pragma unroll
  for (int r = 0; r < 8; ++r) rowsOut[r] = perm[tile * 16 + hi16 * 8 + r];

  const int nBase = strip * (NT * 16);
  const int kbA = hi16 * 16;                  // A contiguous K base (matches B)
  const int kbB = hi16 * 16;                  // B contiguous K base (32x16 layout)
  // Padding lanes (rowA < 0) read from the zero-filled scratch row: one
  // address select outside the loop instead of 16 cndmasks per K-step.
  const _Float16* aRow = (rowA >= 0) ? (A + (size_t)rowA * lda + aOff) : zrow;
  const _Float16* wBase = Wt + ((size_t)e * N + nBase + col16) * K + kbB;

  v8f acc[NT] = {};

  for (int k0 = 0; k0 < K; k0 += 32) {
    v8h lo = *(const v8h*)(aRow + k0 + kbA);
    v8h hh = *(const v8h*)(aRow + k0 + kbA + 8);
    v16h af;
#pragma unroll
    for (int i = 0; i < 8; ++i) { af[i] = lo[i]; af[i + 8] = hh[i]; }
#pragma unroll
    for (int nt = 0; nt < NT; ++nt) {
      const v16h bf = *(const v16h*)(wBase + (size_t)nt * 16 * K + k0);
      acc[nt] = __builtin_amdgcn_wmma_f32_16x16x32_f16(
          /*neg_a=*/false, af, /*neg_b=*/false, bf,
          /*c_mod=*/(short)0, acc[nt], /*reuse_a=*/false, /*reuse_b=*/false);
    }
  }

  // Epilogue: bias + relu + scatter to original row order (coalesced per r:
  // lanes 0-15 / 16-31 write 16 contiguous columns of one row).
#pragma unroll
  for (int nt = 0; nt < NT; ++nt) {
    const int n0 = nBase + nt * 16;
    const float bv = bias[(size_t)e * N + n0 + col16];
#pragma unroll
    for (int r = 0; r < 8; ++r) {
      const int row = rowsOut[r];
      if (row < 0) continue;                  // padding slot
      float v = acc[nt][r] + bv;
      if (RELU && v < 0.0f) v = 0.0f;
      const size_t o = (size_t)row * ldo + oOff + n0 + col16;
      if constexpr (F32OUT) outF[o] = v;
      else                  outH[o] = (_Float16)v;
    }
  }
}

template<int K, int N, int NT, bool RELU, bool F32OUT>
static inline void launch_gemm(const _Float16* A, int lda, int aOff,
                               const _Float16* Wt, const float* bias,
                               const int* perm, const int* starts, const _Float16* zrow,
                               _Float16* outH, float* outF, int ldo, int oOff,
                               hipStream_t stream) {
  constexpr int NSTRIP = N / (NT * 16);
  const int blocks = (MAXTILES * NSTRIP + 7) / 8;
  k_gemm<K, N, NT, RELU, F32OUT><<<blocks, 256, 0, stream>>>(
      A, lda, aOff, Wt, bias, perm, starts, zrow, outH, outF, ldo, oOff);
}

extern "C" void kernel_launch(void* const* d_in, const int* in_sizes, int n_in,
                              void* d_out, int out_size, void* d_ws, size_t ws_size,
                              hipStream_t stream) {
  (void)in_sizes; (void)n_in; (void)out_size; (void)ws_size;

  const float* input = (const float*)d_in[0];
  const float* W00 = (const float*)d_in[1];
  const float* b00 = (const float*)d_in[2];
  const float* W01 = (const float*)d_in[3];
  const float* b01 = (const float*)d_in[4];
  const float* W1p = (const float*)d_in[5];
  const float* b1p = (const float*)d_in[6];
  const float* W1a = (const float*)d_in[7];
  const float* b1a = (const float*)d_in[8];
  const float* W1o = (const float*)d_in[9];
  const float* b1o = (const float*)d_in[10];
  float* out = (float*)d_out;

  char* ws = (char*)d_ws;
  size_t off = 0;
  auto carve = [&](size_t bytes) -> void* {
    void* p = ws + off;
    off += (bytes + 255) & ~(size_t)255;
    return p;
  };
  _Float16* w00t = (_Float16*)carve((size_t)NEXP *  64 * 256 * 2);
  _Float16* w01t = (_Float16*)carve((size_t)NEXP * 256 * 128 * 2);
  _Float16* w1pt = (_Float16*)carve((size_t)NEXP *  64 * 256 * 2);
  _Float16* w1at = (_Float16*)carve((size_t)NEXP * 384 * 128 * 2);
  _Float16* w1ot = (_Float16*)carve((size_t)NEXP * 128 *  32 * 2);
  _Float16* featF16 = (_Float16*)carve((size_t)BATCH * 128 * 2);
  _Float16* act1  = (_Float16*)carve((size_t)BATCH * 256 * 2);   // layer1 out (H=256)
  _Float16* act23 = (_Float16*)carve((size_t)BATCH * 384 * 2);   // [h(128) | g(256)]
  _Float16* act4  = (_Float16*)carve((size_t)BATCH * 128 * 2);   // layer4 out (O0=128)
  _Float16* zrow  = (_Float16*)carve((size_t)ZROWLEN * 2);       // zero A-row for padding
  int* perm0 = (int*)carve((size_t)MAXSLOTS * 4);
  int* perm1 = (int*)carve((size_t)MAXSLOTS * 4);
  int* e0a = (int*)carve((size_t)BATCH * 4);
  int* e1a = (int*)carve((size_t)BATCH * 4);
  int* r0a = (int*)carve((size_t)BATCH * 4);
  int* r1a = (int*)carve((size_t)BATCH * 4);
  int* cnt = (int*)carve(64);
  int* s0  = (int*)carve(64);
  int* s1  = (int*)carve(64);

  const dim3 blk(256);

  // Prep + routing
  k_init<<<(MAXSLOTS + 255) / 256, blk, 0, stream>>>(cnt, perm0, perm1, zrow);
  k_feat<<<(BATCH * 128 + 255) / 256, blk, 0, stream>>>(input, featF16);
  k_wconv<<<(NEXP *  64 * 256 + 255) / 256, blk, 0, stream>>>(W00, w00t,  64, 256);
  k_wconv<<<(NEXP * 256 * 128 + 255) / 256, blk, 0, stream>>>(W01, w01t, 256, 128);
  k_wconv<<<(NEXP *  64 * 256 + 255) / 256, blk, 0, stream>>>(W1p, w1pt,  64, 256);
  k_wconv<<<(NEXP * 384 * 128 + 255) / 256, blk, 0, stream>>>(W1a, w1at, 384, 128);
  k_wconv<<<(NEXP * 128 *  32 + 255) / 256, blk, 0, stream>>>(W1o, w1ot, 128,  32);
  k_route<<<(BATCH + 255) / 256, blk, 0, stream>>>(input, e0a, e1a, r0a, r1a, cnt);
  k_offsets<<<1, 32, 0, stream>>>(cnt, s0, s1);
  k_scatter<<<(BATCH + 255) / 256, blk, 0, stream>>>(e0a, e1a, r0a, r1a, s0, s1, perm0, perm1);

  // Layer 1: feat0(64) -> H=256, expert e0, relu          -> act1
  launch_gemm< 64, 256, 4, true,  false>(featF16, 128,  0, w00t, b00, perm0, s0, zrow,
                                         act1,  nullptr, 256,   0, stream);
  // Layer 2: act1(256) -> O0=128, expert e0, relu         -> act23[:, 0:128]
  launch_gemm<256, 128, 4, true,  false>(act1,  256,  0, w01t, b01, perm0, s0, zrow,
                                         act23, nullptr, 384,   0, stream);
  // Layer 3: feat1(64) -> H=256, expert e1, relu          -> act23[:, 128:384]
  launch_gemm< 64, 256, 4, true,  false>(featF16, 128, 64, w1pt, b1p, perm1, s1, zrow,
                                         act23, nullptr, 384, 128, stream);
  // Layer 4: concat(384) -> O0=128, expert e1, relu       -> act4
  launch_gemm<384, 128, 4, true,  false>(act23, 384,  0, w1at, b1a, perm1, s1, zrow,
                                         act4,  nullptr, 128,   0, stream);
  // Layer 5: act4(128) -> O1=32, expert e1, no relu, fp32 -> d_out
  launch_gemm<128,  32, 2, false, true >(act4,  128,  0, w1ot, b1o, perm1, s1, zrow,
                                         nullptr,  out,  32,   0, stream);
}
